// MultiHeadAttention_17995912970715
// MI455X (gfx1250) — matrix operand order repaired and checked by reading, not compiled, not run on hardware
//
#include <hip/hip_runtime.h>
#include <hip/hip_bf16.h>

// ---------------------------------------------------------------------------
// MHA forward for MI455X (gfx1250, wave32, WMMA).
// Shapes: B=4, T=2048, D=1024, H=16, Dh=64. SCALE = 1/sqrt(1024) = 0.03125.
// Outputs: y (B*T*D f32) then att (B*H*T*T f32), concatenated in d_out.
// ---------------------------------------------------------------------------

typedef __attribute__((ext_vector_type(16))) _Float16 v16h;
typedef __attribute__((ext_vector_type(8)))  _Float16 v8h;
typedef __attribute__((ext_vector_type(8)))  float    v8f;
typedef __attribute__((ext_vector_type(4)))  float    v4f;

constexpr int kB  = 4;
constexpr int kT  = 2048;
constexpr int kD  = 1024;
constexpr int kH  = 16;
constexpr int kDh = 64;
constexpr float kScale = 0.03125f;      // 1/sqrt(1024)
constexpr float kNegInf = -1.0e30f;

__device__ inline v8f wmma16(v16h a, v16h b, v8f c) {
  // v_wmma_f32_16x16x32_f16: D = A(16x32) * B(32x16) + C(16x16 f32)
  return __builtin_amdgcn_wmma_f32_16x16x32_f16(
      /*neg_a=*/false, a, /*neg_b=*/false, b,
      /*c_mod=*/(short)0, c, /*reuse_a=*/false, /*reuse_b=*/false);
}

__device__ inline float waveMax16(float v) {
  v = fmaxf(v, __shfl_xor(v, 1));
  v = fmaxf(v, __shfl_xor(v, 2));
  v = fmaxf(v, __shfl_xor(v, 4));
  v = fmaxf(v, __shfl_xor(v, 8));
  return v;
}
__device__ inline float waveSum16(float v) {
  v += __shfl_xor(v, 1);
  v += __shfl_xor(v, 2);
  v += __shfl_xor(v, 4);
  v += __shfl_xor(v, 8);
  return v;
}

// ---------------------------------------------------------------------------
// Kernel 1: fused QKV projections.  Out = X @ W^T, X (8192x1024 f32),
// W (1024x1024 f32), Out f16.  blockIdx.y in {0,1,2} selects Q/K/V.
// One wave per 16x16 output tile; 4 waves per block.
// ---------------------------------------------------------------------------
__global__ __launch_bounds__(128)
void qkv_proj_kernel(const float* __restrict__ Qin, const float* __restrict__ Kin,
                     const float* __restrict__ Vin,
                     const float* __restrict__ Wq, const float* __restrict__ Wk,
                     const float* __restrict__ Wv,
                     _Float16* __restrict__ qb, _Float16* __restrict__ kb,
                     _Float16* __restrict__ vb) {
  const int tid  = threadIdx.x;
  const int wave = tid >> 5;
  const int lane = tid & 31;
  const int hh   = lane >> 4;   // half-wave: 0 or 1
  const int l15  = lane & 15;

  const float* X; const float* W; _Float16* O;
  if (blockIdx.y == 0)      { X = Qin; W = Wq; O = qb; }
  else if (blockIdx.y == 1) { X = Kin; W = Wk; O = kb; }
  else                      { X = Vin; W = Wv; O = vb; }

  const int tile = blockIdx.x * 4 + wave;        // 512*64 = 32768 tiles
  const int tm = tile >> 6, tn = tile & 63;
  const int row0 = tm * 16, n0 = tn * 16;

  const float* arow = X + (size_t)(row0 + l15) * kD;   // A row for this lane
  const float* brow = W + (size_t)(n0  + l15) * kD;    // W row n -> B column n

  v8f acc = {0.f,0.f,0.f,0.f,0.f,0.f,0.f,0.f};
  for (int k0 = 0; k0 < kD; k0 += 32) {
    // A tile (16x32 f16): element i -> K = (i>>3)*16 + hh*8 + (i&7)
    const float* ap = arow + k0 + hh * 8;
    v4f f0 = *(const v4f*)(ap);
    v4f f1 = *(const v4f*)(ap + 4);
    v4f f2 = *(const v4f*)(ap + 16);
    v4f f3 = *(const v4f*)(ap + 20);
    // B tile (32x16 f16): element i -> K = (i>>3)*16 + hh*8 + (i&7), N = l15
    const float* bp = brow + k0 + hh * 8;
    v4f g0 = *(const v4f*)(bp);
    v4f g1 = *(const v4f*)(bp + 4);
    v4f g2 = *(const v4f*)(bp + 16);
    v4f g3 = *(const v4f*)(bp + 20);
    v16h a, bmat;
#pragma unroll
    for (int i = 0; i < 4; ++i) {
      a[i]      = (_Float16)f0[i];
      a[4 + i]  = (_Float16)f1[i];
      a[8 + i]  = (_Float16)f2[i];
      a[12 + i] = (_Float16)f3[i];
      bmat[i]      = (_Float16)g0[i];
      bmat[4 + i]  = (_Float16)g1[i];
      bmat[8 + i]  = (_Float16)g2[i];
      bmat[12 + i] = (_Float16)g3[i];
    }
    acc = wmma16(a, bmat, acc);
  }
  // D layout: element t -> M = t + 8*hh, N = l15
#pragma unroll
  for (int t = 0; t < 8; ++t)
    O[(size_t)(row0 + t + 8 * hh) * kD + n0 + l15] = (_Float16)acc[t];
}

// ---------------------------------------------------------------------------
// Kernel 2: attention per (b, h, 16-query-row tile).  4 waves per block;
// each wave owns column tiles j = wave, wave+4, ...  Online softmax stats
// (pass 1) -> raw masked scores stored to att; pass 2 normalizes in place
// and WMMA-accumulates y = P @ v.
// ---------------------------------------------------------------------------
__global__ __launch_bounds__(128)
void attn_kernel(const _Float16* __restrict__ qb, const _Float16* __restrict__ kb,
                 const _Float16* __restrict__ vb,
                 float* __restrict__ att, _Float16* __restrict__ yb) {
  const int tid  = threadIdx.x;
  const int wave = tid >> 5;
  const int lane = tid & 31;
  const int hh   = lane >> 4;
  const int l15  = lane & 15;
  const int r = blockIdx.x;        // query row tile (0..127)
  const int h = blockIdx.y;        // head
  const int b = blockIdx.z;        // batch
  const int row0 = 16 * r;

  float* attBase = att + (size_t)(b * kH + h) * kT * kT;

  __shared__ float pmax[4][16];
  __shared__ float psum[4][16];
  __shared__ float rmax[16];
  __shared__ float rinv[16];
  __shared__ float ysm[16 * 64];

  // ---- zero-fill fully masked upper-triangle columns [16(r+1), 2048) ----
  {
    const int c0 = row0 + 16;
    if (c0 < kT) {
      const int cnt4 = (kT - c0) >> 2;
      const v4f z = {0.f, 0.f, 0.f, 0.f};
      for (int i = tid; i < 16 * cnt4; i += 128) {
        const int rr = i / cnt4;
        const int cc = c0 + (i - rr * cnt4) * 4;
        *(v4f*)(attBase + (size_t)(row0 + rr) * kT + cc) = z;
      }
    }
  }

  // ---- preload q A-operand tiles (reused across all column tiles) ----
  const _Float16* qrow = qb + ((size_t)(b * kT) + row0 + l15) * kD + h * kDh;
  v16h aQ[2];
#pragma unroll
  for (int t = 0; t < 2; ++t) {
    const int d0 = t * 32;
    v8h g0 = *(const v8h*)(qrow + d0 + hh * 16);
    v8h g1 = *(const v8h*)(qrow + d0 + hh * 16 + 8);
#pragma unroll
    for (int i = 0; i < 8; ++i) { aQ[t][i] = g0[i]; aQ[t][8 + i] = g1[i]; }
  }

  // ---- pass 1: raw masked scores + online per-row max/sum ----
  float vmax[8], vsum[8];
#pragma unroll
  for (int t = 0; t < 8; ++t) { vmax[t] = kNegInf; vsum[t] = 0.f; }

  for (int j = wave; j <= r; j += 4) {
    const _Float16* krow = kb + ((size_t)(b * kT) + j * 16 + l15) * kD + h * kDh;
    v16h b0 = *(const v16h*)(krow + hh * 16);
    v16h b1 = *(const v16h*)(krow + 32 + hh * 16);
    v8f acc = {0.f,0.f,0.f,0.f,0.f,0.f,0.f,0.f};
    acc = wmma16(aQ[0], b0, acc);
    acc = wmma16(aQ[1], b1, acc);
    const int col = j * 16 + l15;
#pragma unroll
    for (int t = 0; t < 8; ++t) {
      const int m = row0 + t + 8 * hh;           // global query row
      float s = acc[t] * kScale;
      if (col > m) s = kNegInf;                  // causal mask
      attBase[(size_t)m * kT + col] = s;         // raw score (overwritten later)
      const float tm = waveMax16(s);             // tile row max
      const float te = __expf(s - tm);
      const float ts = waveSum16(te);            // tile row sum
      const float nm = fmaxf(vmax[t], tm);
      vsum[t] = vsum[t] * __expf(vmax[t] - nm) + ts * __expf(tm - nm);
      vmax[t] = nm;
    }
  }
  if (l15 == 0) {
#pragma unroll
    for (int t = 0; t < 8; ++t) {
      pmax[wave][t + 8 * hh] = vmax[t];
      psum[wave][t + 8 * hh] = vsum[t];
    }
  }
  __threadfence_block();          // make raw-score global stores visible in-block
  __syncthreads();

  if (tid < 16) {
    float m = fmaxf(fmaxf(pmax[0][tid], pmax[1][tid]),
                    fmaxf(pmax[2][tid], pmax[3][tid]));
    float s = 0.f;
#pragma unroll
    for (int w = 0; w < 4; ++w) s += psum[w][tid] * __expf(pmax[w][tid] - m);
    rmax[tid] = m;
    rinv[tid] = 1.f / s;
  }
  for (int i = tid; i < 16 * 64; i += 128) ysm[i] = 0.f;
  __syncthreads();

  // ---- pass 2: normalize in place and accumulate y = P @ v ----
  const float mA   = rmax[l15];      // A-layout: row M == l15 for this lane
  const float invA = rinv[l15];
  const int   mrow = row0 + l15;     // global query row for this lane's A rows
  v8f accY[4];
#pragma unroll
  for (int t = 0; t < 4; ++t) accY[t] = (v8f){0.f,0.f,0.f,0.f,0.f,0.f,0.f,0.f};

  const int nchunks = (r + 2) >> 1;  // ceil((r+1)/2) chunks of 32 columns
  for (int c = wave; c < nchunks; c += 4) {
    const int colbase = c * 32;
    v16h aP;
#pragma unroll
    for (int i = 0; i < 16; ++i) {
      const int K = hh * 16 + i;
      const int col = colbase + K;
      float p = 0.f;
      if (col <= mrow) {
        const float s = attBase[(size_t)mrow * kT + col];
        p = __expf(s - mA) * invA;
      }
      attBase[(size_t)mrow * kT + col] = p;   // final normalized att
      aP[i] = (_Float16)p;
    }
    const _Float16* vcol =
        vb + ((size_t)(b * kT) + colbase + hh * 16) * kD + h * kDh + l15;
#pragma unroll
    for (int t = 0; t < 4; ++t) {
      v16h bV;
#pragma unroll
      for (int i = 0; i < 16; ++i) bV[i] = vcol[(size_t)i * kD + t * 16];
      accY[t] = wmma16(aP, bV, accY[t]);
    }
  }
  // cross-wave reduce into LDS (ds_add_f32)
#pragma unroll
  for (int t = 0; t < 4; ++t)
#pragma unroll
    for (int e = 0; e < 8; ++e)
      atomicAdd(&ysm[(e + 8 * hh) * 64 + t * 16 + l15], accY[t][e]);
  __syncthreads();

  for (int i = tid; i < 16 * 64; i += 128) {
    const int rr = i >> 6, cc = i & 63;
    yb[((size_t)(b * kT) + row0 + rr) * kD + h * kDh + cc] = (_Float16)ysm[i];
  }
}

// ---------------------------------------------------------------------------
// Kernel 3: out = y @ Wproj^T.  y f16 (8192x1024), Wproj f32, out f32.
// ---------------------------------------------------------------------------
__global__ __launch_bounds__(128)
void out_proj_kernel(const _Float16* __restrict__ yb,
                     const float* __restrict__ Wp, float* __restrict__ out) {
  const int tid  = threadIdx.x;
  const int wave = tid >> 5;
  const int lane = tid & 31;
  const int hh   = lane >> 4;
  const int l15  = lane & 15;

  const int tile = blockIdx.x * 4 + wave;
  const int tm = tile >> 6, tn = tile & 63;
  const int row0 = tm * 16, n0 = tn * 16;

  const _Float16* arow = yb + (size_t)(row0 + l15) * kD;
  const float*    brow = Wp + (size_t)(n0 + l15) * kD;

  v8f acc = {0.f,0.f,0.f,0.f,0.f,0.f,0.f,0.f};
  for (int k0 = 0; k0 < kD; k0 += 32) {
    v8h g0 = *(const v8h*)(arow + k0 + hh * 16);
    v8h g1 = *(const v8h*)(arow + k0 + hh * 16 + 8);
    const float* bp = brow + k0 + hh * 16;
    v4f w0 = *(const v4f*)(bp);
    v4f w1 = *(const v4f*)(bp + 4);
    v4f w2 = *(const v4f*)(bp + 8);
    v4f w3 = *(const v4f*)(bp + 12);
    v16h a, bmat;
#pragma unroll
    for (int i = 0; i < 8; ++i) { a[i] = g0[i]; a[8 + i] = g1[i]; }
#pragma unroll
    for (int i = 0; i < 4; ++i) {
      bmat[i]      = (_Float16)w0[i];
      bmat[4 + i]  = (_Float16)w1[i];
      bmat[8 + i]  = (_Float16)w2[i];
      bmat[12 + i] = (_Float16)w3[i];
    }
    acc = wmma16(a, bmat, acc);
  }
#pragma unroll
  for (int t = 0; t < 8; ++t)
    out[(size_t)(row0 + t + 8 * hh) * kD + n0 + l15] = acc[t];
}

// ---------------------------------------------------------------------------
extern "C" void kernel_launch(void* const* d_in, const int* in_sizes, int n_in,
                              void* d_out, int out_size, void* d_ws, size_t ws_size,
                              hipStream_t stream) {
  (void)in_sizes; (void)n_in; (void)out_size; (void)ws_size;

  const float* Qin   = (const float*)d_in[0];
  const float* Kin   = (const float*)d_in[1];
  const float* Vin   = (const float*)d_in[2];
  const float* Wq    = (const float*)d_in[3];
  const float* Wk    = (const float*)d_in[4];
  const float* Wv    = (const float*)d_in[5];
  const float* Wproj = (const float*)d_in[6];

  float* out_y   = (float*)d_out;
  float* out_att = out_y + (size_t)kB * kT * kD;   // att follows y

  // Workspace: q, k, v, y as f16 (each 8192*1024 halfs = 16 MiB; 64 MiB total)
  const size_t elems = (size_t)kB * kT * kD;
  _Float16* qb = (_Float16*)d_ws;
  _Float16* kb = qb + elems;
  _Float16* vb = kb + elems;
  _Float16* yb = vb + elems;

  const dim3 blk(128);
  // 512 row-tiles * 64 col-tiles = 32768 tiles, 4 tiles per block
  qkv_proj_kernel<<<dim3(8192, 3), blk, 0, stream>>>(Qin, Kin, Vin, Wq, Wk, Wv,
                                                     qb, kb, vb);
  attn_kernel<<<dim3(kT / 16, kH, kB), blk, 0, stream>>>(qb, kb, vb, out_att, yb);
  out_proj_kernel<<<dim3(8192), blk, 0, stream>>>(yb, Wproj, out_y);
}
